// SparseConv3D_20194936226218
// MI455X (gfx1250) — matrix operand order repaired and checked by reading, not compiled, not run on hardware
//
#include <hip/hip_runtime.h>
#include <hip/hip_bf16.h>
#include <stdint.h>

typedef __attribute__((ext_vector_type(8)))  __bf16 v8bf;
typedef __attribute__((ext_vector_type(16))) __bf16 v16bf;
typedef __attribute__((ext_vector_type(8)))  float  v8f;

#define N_VOX   131072
#define M_IDX   65536
#define C_IN    256
#define C_OUT   256
#define K_OFF   27

#define ROWS    64            // gathered rows per block (4 row-tiles of 16)
#define RT      (ROWS / 16)   // row tiles per block = 4
#define APAD    264           // 256 + 8 bf16 pad -> 528B row stride, conflict-free b128

// ---------------------------------------------------------------------------
// Kernel 1: zero the output accumulator (harness poisons d_out; atomics add)
// ---------------------------------------------------------------------------
__global__ __launch_bounds__(256) void zero_out(float* __restrict__ out, int total) {
    int i = blockIdx.x * 256 + threadIdx.x;
    if (i < total) out[i] = 0.0f;
}

// ---------------------------------------------------------------------------
// Kernel 2: convert inputs fp32 -> bf16 (row-major [N, C_IN])
// ---------------------------------------------------------------------------
__global__ __launch_bounds__(256) void convert_inputs(const float* __restrict__ x,
                                                      __bf16* __restrict__ xb, int total) {
    int i = blockIdx.x * 256 + threadIdx.x;
    if (i < total) xb[i] = (__bf16)x[i];
}

// ---------------------------------------------------------------------------
// Kernel 3: pre-swizzle weights into per-lane WMMA B-fragment order.
// Layout: Wsw[((k*8 + s)*16 + t)*512 + lane*16 + i]
//   B is K(32) x N(16): lanes 0-15 hold K=0..15, lanes 16-31 hold K=16..31;
//   element i is the K index inside the lane's half.
//   col = t*16 + (lane & 15);  cin = s*32 + (lane >> 4)*16 + i
// ---------------------------------------------------------------------------
__global__ __launch_bounds__(256) void swizzle_weights(const float* __restrict__ w,
                                                       __bf16* __restrict__ wsw) {
    int bid = blockIdx.x;              // 0 .. 27*8*16-1
    int k = bid >> 7;
    int s = (bid >> 4) & 7;
    int t = bid & 15;
    for (int e = threadIdx.x; e < 512; e += 256) {
        int lane = e >> 4;
        int i    = e & 15;
        int col  = t * 16 + (lane & 15);
        int cin  = s * 32 + (lane >> 4) * 16 + i;
        float v  = w[((size_t)k * C_IN + cin) * C_OUT + col];
        wsw[(size_t)bid * 512 + e] = (__bf16)v;
    }
}

// ---------------------------------------------------------------------------
// Kernel 4: main gather -> WMMA GEMM -> scatter-add.
// Block: 256 threads = 8 wave32. blockIdx.y = kernel offset k,
// blockIdx.x = 64-row tile of the M=65536 gathered rows.
// Wave w owns cout tiles (2w, 2w+1) across all 4 row-tiles:
//   8 accumulators (16x16 f32 each) -> 64 WMMAs per wave.
// B fragments are loaded once per K-step and reused across 4 row-tiles.
// ---------------------------------------------------------------------------
__global__ __launch_bounds__(256) void spconv_wmma(const __bf16* __restrict__ xb,
                                                   const __bf16* __restrict__ wsw,
                                                   const int*    __restrict__ idx,
                                                   float*        __restrict__ out) {
    __shared__ __bf16 sA[ROWS * APAD];   // 64 rows x 264 bf16 = 33792 B
    __shared__ int    sIdx[ROWS];

    const int k    = blockIdx.y;
    const int tile = blockIdx.x;
    const int tid  = threadIdx.x;
    const int lane = tid & 31;
    const int wave = tid >> 5;     // 0..7
    const int half = lane >> 4;    // 0 or 1
    const int l16  = lane & 15;

    if (tid < ROWS)
        sIdx[tid] = idx[(size_t)k * M_IDX + (size_t)tile * ROWS + tid];
    __syncthreads();

    // Stage 64 gathered rows (bf16) into LDS: 1024 chunks of 32B, 4 per thread.
#pragma unroll
    for (int j = 0; j < 4; ++j) {
        int c     = tid + 256 * j;     // 0..1023
        int row   = c >> 4;            // 0..63
        int chunk = c & 15;            // 16 bf16 each
        const v8bf* src = (const v8bf*)(xb + (size_t)sIdx[row] * C_IN + chunk * 16);
        v8bf* dst = (v8bf*)(&sA[row * APAD + chunk * 16]);
        dst[0] = src[0];
        dst[1] = src[1];
    }
    __syncthreads();

    v8f acc[RT][2];
#pragma unroll
    for (int r = 0; r < RT; ++r) {
        acc[r][0] = (v8f){};
        acc[r][1] = (v8f){};
    }

    const int t0 = wave * 2;
    const int t1 = wave * 2 + 1;
    const __bf16* wbase = wsw + (size_t)k * 8 * 16 * 512;

#pragma unroll
    for (int s = 0; s < 8; ++s) {
        // B fragments: contiguous 32B per lane thanks to the pre-swizzle.
        const v16bf b0 = *(const v16bf*)(wbase + (((size_t)s * 16 + t0) * 32 + lane) * 16);
        const v16bf b1 = *(const v16bf*)(wbase + (((size_t)s * 16 + t1) * 32 + lane) * 16);

#pragma unroll
        for (int rt = 0; rt < RT; ++rt) {
            // A fragment (16x32 bf16), same contiguous k order as B:
            //   half h, element i:  K = s*32 + h*16 + i
            const __bf16* ap = &sA[(rt * 16 + l16) * APAD + s * 32 + half * 16];
            v8bf alo = *(const v8bf*)ap;
            v8bf ahi = *(const v8bf*)(ap + 8);
            v16bf a;
#pragma unroll
            for (int i = 0; i < 8; ++i) { a[i] = alo[i]; a[i + 8] = ahi[i]; }

            acc[rt][0] = __builtin_amdgcn_wmma_f32_16x16x32_bf16(false, a, false, b0,
                                                                 (short)0, acc[rt][0],
                                                                 false, false);
            acc[rt][1] = __builtin_amdgcn_wmma_f32_16x16x32_bf16(false, a, false, b1,
                                                                 (short)0, acc[rt][1],
                                                                 false, false);
        }
    }

    // Scatter-add: C/D layout => VGPR r holds M = r + 8*half, N = l16.
#pragma unroll
    for (int rt = 0; rt < RT; ++rt) {
#pragma unroll
        for (int r = 0; r < 8; ++r) {
            int m   = rt * 16 + r + half * 8;
            int row = sIdx[m];
            atomicAdd(&out[(size_t)row * C_OUT + t0 * 16 + l16], acc[rt][0][r]);
            atomicAdd(&out[(size_t)row * C_OUT + t1 * 16 + l16], acc[rt][1][r]);
        }
    }
}

// ---------------------------------------------------------------------------
// Kernel 5: bias + ReLU epilogue
// ---------------------------------------------------------------------------
__global__ __launch_bounds__(256) void bias_relu(float* __restrict__ out,
                                                 const float* __restrict__ bias, int total) {
    int i = blockIdx.x * 256 + threadIdx.x;
    if (i < total) {
        float v = out[i] + bias[i & (C_OUT - 1)];
        out[i] = v > 0.0f ? v : 0.0f;
    }
}

// ---------------------------------------------------------------------------
extern "C" void kernel_launch(void* const* d_in, const int* in_sizes, int n_in,
                              void* d_out, int out_size, void* d_ws, size_t ws_size,
                              hipStream_t stream) {
    const float* x    = (const float*)d_in[0];   // [N, 256] fp32
    const float* w    = (const float*)d_in[1];   // [3,3,3,256,256] fp32 == [27,256,256]
    const float* bias = (const float*)d_in[2];   // [1,256] fp32
    // d_in[3] = voxel_idx (unused by the reference computation)
    const int*   idx  = (const int*)d_in[4];     // [27, 65536, 1] int32
    float* out = (float*)d_out;

    // Workspace: bf16 inputs (64 MiB) then swizzled bf16 weights (3.375 MiB)
    __bf16* xb  = (__bf16*)d_ws;
    __bf16* wsw = (__bf16*)((char*)d_ws + (size_t)N_VOX * C_IN * sizeof(__bf16));

    const int total_out = N_VOX * C_OUT;         // 33,554,432
    const int total_in  = N_VOX * C_IN;          // 33,554,432

    zero_out<<<(total_out + 255) / 256, 256, 0, stream>>>(out, total_out);
    convert_inputs<<<(total_in + 255) / 256, 256, 0, stream>>>(x, xb, total_in);
    swizzle_weights<<<K_OFF * 8 * 16, 256, 0, stream>>>(w, wsw);

    dim3 grid(M_IDX / ROWS, K_OFF);              // 1024 x 27 blocks
    spconv_wmma<<<grid, 256, 0, stream>>>(xb, wsw, idx, out);

    bias_relu<<<(total_out + 255) / 256, 256, 0, stream>>>(out, bias, total_out);
}
